// NodeGenerator_85856396247142
// MI455X (gfx1250) — matrix operand order repaired and checked by reading, not compiled, not run on hardware
//
#include <hip/hip_runtime.h>
#include <math.h>

// ---------------------------------------------------------------------------
// CDNA5 (gfx1250) wave32 WMMA types
// ---------------------------------------------------------------------------
typedef __attribute__((ext_vector_type(16))) __bf16   v16bf;
typedef __attribute__((ext_vector_type(8)))  float    v8f;
typedef __attribute__((ext_vector_type(4)))  unsigned v4u;

union BF16x16 {
    v16bf bf;
    v4u   u4[2];
};

// Native device-scope f32 atomic add (non-returning -> STOREcnt path, RMW in L2).
__device__ __forceinline__ void atomic_add_f32_dev(float* addr, float val) {
    asm volatile("global_atomic_add_f32 %0, %1, off scope:SCOPE_DEV"
                 :: "v"(addr), "v"(val) : "memory");
}

// Async global -> LDS 16B copy (CDNA5, tracked by ASYNCcnt).
__device__ __forceinline__ void async_g2lds_b128(unsigned ldsAddr, const void* gaddr) {
    asm volatile("global_load_async_to_lds_b128 %0, %1, off"
                 :: "v"(ldsAddr), "v"(gaddr) : "memory");
}
__device__ __forceinline__ void wait_asynccnt0() {
    asm volatile("s_wait_asynccnt 0x0" ::: "memory");
}
__device__ __forceinline__ unsigned lds_addr_of(const void* shared_ptr) {
    // Generic LDS address = {SHARED_BASE, offset}; low 32 bits are the LDS byte address.
    return (unsigned)(size_t)shared_ptr;
}

// ---------------------------------------------------------------------------
// Kernel 0: zero a float buffer (deg / nsum init; harness does not zero ws)
// ---------------------------------------------------------------------------
__global__ void zero_f32(float* __restrict__ p, size_t n) {
    size_t i = (size_t)blockIdx.x * blockDim.x + threadIdx.x;
    size_t stride = (size_t)gridDim.x * blockDim.x;
    for (; i < n; i += stride) p[i] = 0.0f;
}

// ---------------------------------------------------------------------------
// Kernel 1: one-shot weight pre-pack: transposed [out n][in k] bf16, padded.
// Layout in wpack (bf16 elements):
//   [0,16384)        W1^T  (128 x 128)
//   [16384,24576)    W2^T  (64  x 128)
//   [24576,29696)    W3^T  padded to 80 rows x 64
//   [29696,31744)    P1^T  (32 x 64)
//   [31744,32256)    P2^T  padded to 16 rows x 32 (row 0 = P2)
// ---------------------------------------------------------------------------
__global__ void pack_weights(const float* __restrict__ W1, const float* __restrict__ W2,
                             const float* __restrict__ W3, const float* __restrict__ P1,
                             const float* __restrict__ P2, __bf16* __restrict__ wp) {
    int i = blockIdx.x * blockDim.x + threadIdx.x;
    if (i < 16384) {
        int n = i >> 7, k = i & 127;
        wp[i] = (__bf16)W1[k * 128 + n];
    } else if (i < 24576) {
        int j = i - 16384; int n = j >> 7, k = j & 127;
        wp[i] = (__bf16)W2[k * 64 + n];
    } else if (i < 29696) {
        int j = i - 24576; int n = j >> 6, k = j & 63;
        wp[i] = (__bf16)((n < 67) ? W3[k * 67 + n] : 0.0f);
    } else if (i < 31744) {
        int j = i - 29696; int n = j >> 6, k = j & 63;
        wp[i] = (__bf16)P1[k * 32 + n];
    } else if (i < 32256) {
        int j = i - 31744; int n = j >> 5, k = j & 31;
        wp[i] = (__bf16)((n == 0) ? P2[k] : 0.0f);
    }
}

// ---------------------------------------------------------------------------
// Kernel 2: edge scatter.  One wave per edge; lane = feature column (2 cols
// per lane).  Gathers are 128B coalesced; atomics land in L2 (nsum 25.6MB
// fits the 192MB device L2).
// ---------------------------------------------------------------------------
__global__ void scatter_edges(const float* __restrict__ feat,
                              const int*   __restrict__ ei,
                              float* __restrict__ deg,
                              float* __restrict__ nsum,
                              int E) {
    int tid    = blockIdx.x * blockDim.x + threadIdx.x;
    int wave   = tid >> 5;
    int lane   = tid & 31;
    int nWaves = (gridDim.x * blockDim.x) >> 5;
    for (int e = wave; e < E; e += nWaves) {
        int s = ei[e];
        int d = ei[E + e];
        const float* fs = feat + (size_t)s * 64;
        const float* fd = feat + (size_t)d * 64;
        float* ns = nsum + (size_t)s * 64;
        float* nd = nsum + (size_t)d * 64;
        atomic_add_f32_dev(&ns[lane],      fd[lane]);
        atomic_add_f32_dev(&ns[lane + 32], fd[lane + 32]);
        atomic_add_f32_dev(&nd[lane],      fs[lane]);
        atomic_add_f32_dev(&nd[lane + 32], fs[lane + 32]);
        if (lane == 0) {
            atomic_add_f32_dev(&deg[s], 1.0f);
            atomic_add_f32_dev(&deg[d], 1.0f);
        }
    }
}

// ---------------------------------------------------------------------------
// Kernel 3: per-node prep.  One wave per node: softmax(ops)[0], mask, nmean,
// materialize ctx = concat(feat, nmean) as bf16 (N x 128) for WMMA.
// ---------------------------------------------------------------------------
__global__ void prep_nodes(const float* __restrict__ feat,
                           const float* __restrict__ ops,
                           const float* __restrict__ deg,
                           const float* __restrict__ nsum,
                           float*  __restrict__ maskf,
                           __bf16* __restrict__ ctx,
                           int N) {
    int tid  = blockIdx.x * blockDim.x + threadIdx.x;
    int node = tid >> 5;
    int g    = tid & 31;
    if (node >= N) return;
    float o0 = ops[node * 4 + 0], o1 = ops[node * 4 + 1];
    float o2 = ops[node * 4 + 2], o3 = ops[node * 4 + 3];
    float mx = fmaxf(fmaxf(o0, o1), fmaxf(o2, o3));
    float e0 = expf(o0 - mx), e1 = expf(o1 - mx), e2 = expf(o2 - mx), e3 = expf(o3 - mx);
    float p0 = e0 / (e0 + e1 + e2 + e3);
    float dg = deg[node];
    if (g == 0) maskf[node] = (p0 > 0.5f && dg > 0.0f) ? 1.0f : 0.0f;
    float inv = 1.0f / fmaxf(dg, 1.0f);
    const float* f  = feat + (size_t)node * 64;
    const float* ns = nsum + (size_t)node * 64;
    __bf16* c = ctx + (size_t)node * 128;
    c[g]      = (__bf16)f[g];
    c[g + 32] = (__bf16)f[g + 32];
    c[g + 64] = (__bf16)(ns[g] * inv);
    c[g + 96] = (__bf16)(ns[g + 32] * inv);
}

// ---------------------------------------------------------------------------
// WMMA tile loaders (layouts per CDNA5 ISA 7.12.2, wave32):
//  A 16x32 bf16 : lane holds row (lane&15); K chunks at kt*32+(lane>>4)*8, +16.
//  B 32x16 bf16 : lane holds col (lane&15) from transposed [n][k] storage;
//                 K chunks at kt*32+(lane>>4)*8, +16 (same as A).
//  C/D 16x16 f32: VGPR r -> row r+(lane>>4)*8, col lane&15.
// ---------------------------------------------------------------------------
__device__ __forceinline__ BF16x16 load_a_tile(const __bf16* rowPtr, int kt, int hi) {
    BF16x16 a;
    const __bf16* p = rowPtr + kt * 32 + hi * 8;
    a.u4[0] = *(const v4u*)p;
    a.u4[1] = *(const v4u*)(p + 16);
    return a;
}
__device__ __forceinline__ BF16x16 load_b_tile(const __bf16* wt, int ldk, int nt, int kt, int lo, int hi) {
    BF16x16 b;
    const __bf16* p = wt + (size_t)(nt * 16 + lo) * ldk + kt * 32 + hi * 8;
    b.u4[0] = *(const v4u*)p;
    b.u4[1] = *(const v4u*)(p + 16);
    return b;
}

// ---------------------------------------------------------------------------
// Kernel 4: layers 1+2.  2 waves/block, 4 row-tiles per wave.
// LDS: packed W1^T|W2^T (48KB, async-copied) + 2x 16x128 bf16 staging (8KB).
// ---------------------------------------------------------------------------
__global__ __launch_bounds__(64) void mlp12_kernel(
    const __bf16* __restrict__ ctx,    // N x 128 bf16
    const __bf16* __restrict__ wpack,  // packed bf16 weights
    const float*  __restrict__ b1,     // 128
    const float*  __restrict__ b2,     // 64
    __bf16* __restrict__ h2,           // N x 64 bf16
    int nTiles) {
    __shared__ __bf16 sW[24576];            // [0,16384)=W1^T, [16384,24576)=W2^T
    __shared__ __bf16 sStage[2][16 * 128];  // per-wave h1 staging (row-major)

    const int tid = threadIdx.x;
    // Stage 48KB of packed weights into LDS with async b128 copies.
    {
        unsigned ldsBase = lds_addr_of(&sW[0]);
        for (int i = tid; i < 24576 / 8; i += 64)      // 3072 x 16B
            async_g2lds_b128(ldsBase + i * 16, wpack + i * 8);
        wait_asynccnt0();
    }
    __syncthreads();

    const __bf16* sW1t = sW;
    const __bf16* sW2t = sW + 16384;
    const int wave = tid >> 5, lane = tid & 31;
    const int lo = lane & 15, hi = lane >> 4;

    // Hoist biases (loop-invariant per lane).
    float bias1[8], bias2[4];
#pragma unroll
    for (int nt = 0; nt < 8; ++nt) bias1[nt] = b1[nt * 16 + lo];
#pragma unroll
    for (int nt = 0; nt < 4; ++nt) bias2[nt] = b2[nt * 16 + lo];

    __bf16* stage = sStage[wave];
    const int baseTile = (blockIdx.x * 2 + wave) * 4;

    for (int t = 0; t < 4; ++t) {
        const int rowTile = baseTile + t;
        if (rowTile >= nTiles) break;
        const int rowBase = rowTile * 16;

        // A tiles for layer 1 straight from global ctx.
        BF16x16 a[4];
        const __bf16* rowPtr = ctx + (size_t)(rowBase + lo) * 128;
#pragma unroll
        for (int kt = 0; kt < 4; ++kt) a[kt] = load_a_tile(rowPtr, kt, hi);

        // Layer 1: 16x128 @ 128x128 -> relu -> stage bf16.
#pragma unroll
        for (int nt = 0; nt < 8; ++nt) {
            v8f acc = {};
#pragma unroll
            for (int kt = 0; kt < 4; ++kt) {
                BF16x16 b = load_b_tile(sW1t, 128, nt, kt, lo, hi);
                acc = __builtin_amdgcn_wmma_f32_16x16x32_bf16(false, a[kt].bf, false, b.bf,
                                                              (short)0, acc, false, false);
            }
            const int col = nt * 16 + lo;
#pragma unroll
            for (int r = 0; r < 8; ++r) {
                float v = fmaxf(acc[r] + bias1[nt], 0.0f);
                stage[(r + hi * 8) * 128 + col] = (__bf16)v;  // D -> row-major re-layout
            }
        }

        // Layer 2: A from staged h1 (same-wave LDS; DScnt waits by compiler).
        BF16x16 a2[4];
#pragma unroll
        for (int kt = 0; kt < 4; ++kt) a2[kt] = load_a_tile(stage + lo * 128, kt, hi);

#pragma unroll
        for (int nt = 0; nt < 4; ++nt) {
            v8f acc = {};
#pragma unroll
            for (int kt = 0; kt < 4; ++kt) {
                BF16x16 b = load_b_tile(sW2t, 128, nt, kt, lo, hi);
                acc = __builtin_amdgcn_wmma_f32_16x16x32_bf16(false, a2[kt].bf, false, b.bf,
                                                              (short)0, acc, false, false);
            }
            const int col = nt * 16 + lo;
#pragma unroll
            for (int r = 0; r < 8; ++r) {
                float v = fmaxf(acc[r] + bias2[nt], 0.0f);
                h2[(size_t)(rowBase + r + hi * 8) * 64 + col] = (__bf16)v;
            }
        }
    }
}

// ---------------------------------------------------------------------------
// Kernel 5: layer 3 (gen) + head P1/P2 + sigmoid + mask -> N x 68 output.
// 4 waves/block, 4 row-tiles per wave.
// LDS: packed W3^T|P1^T|P2^T (15KB, async-copied) + per-wave staging (12KB).
// ---------------------------------------------------------------------------
__global__ __launch_bounds__(128) void head_kernel(
    const __bf16* __restrict__ h2,     // N x 64 bf16
    const __bf16* __restrict__ wpack,  // packed bf16 weights (head region)
    const float*  __restrict__ b3,     // 67
    const float*  __restrict__ pb1,    // 32
    const float*  __restrict__ pb2,    // 1
    const float*  __restrict__ maskf,  // N
    float* __restrict__ out,           // N x 68
    int nTiles) {
    __shared__ __bf16 sWH[7680];        // W3^T(5120) | P1^T(2048) | P2^T(512)
    __shared__ __bf16 sF[4][16 * 64];   // per-wave feats staging
    __shared__ __bf16 sP[4][16 * 32];   // per-wave p staging

    const int tid = threadIdx.x;
    {
        const __bf16* src = wpack + 24576;   // head region of the pack
        unsigned ldsBase = lds_addr_of(&sWH[0]);
        for (int i = tid; i < 7680 / 8; i += 128)    // 960 x 16B
            async_g2lds_b128(ldsBase + i * 16, src + i * 8);
        wait_asynccnt0();
    }
    __syncthreads();

    const __bf16* sW3t = sWH;
    const __bf16* sP1t = sWH + 5120;
    const __bf16* sP2t = sWH + 5120 + 2048;
    const int wave = tid >> 5, lane = tid & 31;
    const int lo = lane & 15, hi = lane >> 4;

    float bias3[5], biasP1[2];
#pragma unroll
    for (int nt = 0; nt < 5; ++nt) {
        const int col = nt * 16 + lo;
        bias3[nt] = (col < 67) ? b3[col] : 0.0f;
    }
#pragma unroll
    for (int nt = 0; nt < 2; ++nt) biasP1[nt] = pb1[nt * 16 + lo];
    const float bb2 = pb2[0];

    __bf16* sf = sF[wave];
    __bf16* sp = sP[wave];
    const int baseTile = (blockIdx.x * 4 + wave) * 4;

    for (int t = 0; t < 4; ++t) {
        const int rowTile = baseTile + t;
        if (rowTile >= nTiles) break;
        const int rowBase = rowTile * 16;

        BF16x16 a[2];
        const __bf16* rowPtr = h2 + (size_t)(rowBase + lo) * 64;
#pragma unroll
        for (int kt = 0; kt < 2; ++kt) a[kt] = load_a_tile(rowPtr, kt, hi);

        // gen = h2 @ W3 + b3   (5 n-tiles, cols 0..79, valid 0..66)
        v8f gen[5];
#pragma unroll
        for (int nt = 0; nt < 5; ++nt) {
            v8f acc = {};
#pragma unroll
            for (int kt = 0; kt < 2; ++kt) {
                BF16x16 b = load_b_tile(sW3t, 64, nt, kt, lo, hi);
                acc = __builtin_amdgcn_wmma_f32_16x16x32_bf16(false, a[kt].bf, false, b.bf,
                                                              (short)0, acc, false, false);
            }
            const int col = nt * 16 + lo;
#pragma unroll
            for (int r = 0; r < 8; ++r) {
                float v = acc[r] + bias3[nt];
                acc[r] = v;
                if (col >= 3 && col < 67)                    // feats = gen[:, 3:67]
                    sf[(r + hi * 8) * 64 + (col - 3)] = (__bf16)v;
            }
            gen[nt] = acc;
        }

        // p = relu(feats @ P1 + pb1)
        BF16x16 af[2];
#pragma unroll
        for (int kt = 0; kt < 2; ++kt) af[kt] = load_a_tile(sf + lo * 64, kt, hi);

#pragma unroll
        for (int nt = 0; nt < 2; ++nt) {
            v8f acc = {};
#pragma unroll
            for (int kt = 0; kt < 2; ++kt) {
                BF16x16 b = load_b_tile(sP1t, 64, nt, kt, lo, hi);
                acc = __builtin_amdgcn_wmma_f32_16x16x32_bf16(false, af[kt].bf, false, b.bf,
                                                              (short)0, acc, false, false);
            }
            const int col = nt * 16 + lo;
#pragma unroll
            for (int r = 0; r < 8; ++r) {
                float v = fmaxf(acc[r] + biasP1[nt], 0.0f);
                sp[(r + hi * 8) * 32 + col] = (__bf16)v;
            }
        }

        // prob = sigmoid(p @ P2 + pb2)  (single WMMA, only col 0 meaningful)
        BF16x16 ap;
        {
            const __bf16* p = sp + lo * 32 + hi * 8;
            ap.u4[0] = *(const v4u*)p;
            ap.u4[1] = *(const v4u*)(p + 16);
        }
        BF16x16 bp;
        {
            const __bf16* p = sP2t + lo * 32 + hi * 8;
            bp.u4[0] = *(const v4u*)p;
            bp.u4[1] = *(const v4u*)(p + 16);
        }
        v8f pacc = {};
        pacc = __builtin_amdgcn_wmma_f32_16x16x32_bf16(false, ap.bf, false, bp.bf,
                                                       (short)0, pacc, false, false);

        float m[8];
#pragma unroll
        for (int r = 0; r < 8; ++r) m[r] = maskf[rowBase + r + hi * 8];

        // out[:, 0:67] = gen[:, 0:67] * mask
#pragma unroll
        for (int nt = 0; nt < 5; ++nt) {
            const int col = nt * 16 + lo;
            if (col < 67) {
#pragma unroll
                for (int r = 0; r < 8; ++r)
                    out[(size_t)(rowBase + r + hi * 8) * 68 + col] = gen[nt][r] * m[r];
            }
        }
        // out[:, 67] = prob * mask  (lane with lo==0 holds col 0 of the D tile)
        if (lo == 0) {
#pragma unroll
            for (int r = 0; r < 8; ++r) {
                float x = pacc[r] + bb2;
                float s = 1.0f / (1.0f + expf(-x));
                out[(size_t)(rowBase + r + hi * 8) * 68 + 67] = s * m[r];
            }
        }
    }
}

// ---------------------------------------------------------------------------
// Host launcher
// ---------------------------------------------------------------------------
static inline size_t align256(size_t x) { return (x + 255) & ~(size_t)255; }

extern "C" void kernel_launch(void* const* d_in, const int* in_sizes, int n_in,
                              void* d_out, int out_size, void* d_ws, size_t ws_size,
                              hipStream_t stream) {
    const float* feat = (const float*)d_in[0];
    const float* ops  = (const float*)d_in[1];
    const int*   ei   = (const int*)d_in[2];
    const float* W1   = (const float*)d_in[3];
    const float* b1   = (const float*)d_in[4];
    const float* W2   = (const float*)d_in[5];
    const float* b2   = (const float*)d_in[6];
    const float* W3   = (const float*)d_in[7];
    const float* b3   = (const float*)d_in[8];
    const float* P1   = (const float*)d_in[9];
    const float* pb1  = (const float*)d_in[10];
    const float* P2   = (const float*)d_in[11];
    const float* pb2  = (const float*)d_in[12];
    float* out = (float*)d_out;

    const int N  = in_sizes[0] / 64;
    const int E  = in_sizes[2] / 2;
    const int NT = (N + 15) / 16;   // N = 100000 -> exactly 6250 tiles

    // Workspace layout
    char* ws = (char*)d_ws;
    size_t off = 0;
    float*  deg   = (float*)(ws + off);  off = align256(off + sizeof(float) * (size_t)N);
    float*  nsum  = (float*)(ws + off);  off = align256(off + sizeof(float) * (size_t)N * 64);
    float*  maskf = (float*)(ws + off);  off = align256(off + sizeof(float) * (size_t)N);
    __bf16* ctx   = (__bf16*)(ws + off); off = align256(off + sizeof(__bf16) * (size_t)N * 128);
    __bf16* h2    = (__bf16*)(ws + off); off = align256(off + sizeof(__bf16) * (size_t)N * 64);
    __bf16* wpack = (__bf16*)(ws + off); off = align256(off + sizeof(__bf16) * (size_t)32256);
    (void)ws_size; (void)n_in; (void)out_size;

    zero_f32<<<1024, 256, 0, stream>>>(deg, (size_t)N);
    zero_f32<<<2048, 256, 0, stream>>>(nsum, (size_t)N * 64);
    pack_weights<<<(32256 + 255) / 256, 256, 0, stream>>>(W1, W2, W3, P1, P2, wpack);
    scatter_edges<<<4096, 256, 0, stream>>>(feat, ei, deg, nsum, E);
    prep_nodes<<<(N * 32 + 255) / 256, 256, 0, stream>>>(feat, ops, deg, nsum, maskf, ctx, N);
    mlp12_kernel<<<(NT + 7) / 8, 64, 0, stream>>>(ctx, wpack, b1, b2, h2, NT);
    head_kernel<<<(NT + 15) / 16, 128, 0, stream>>>(h2, wpack, b3, pb1, pb2, maskf, out, NT);
}
